// FilterAttention_88837103550623
// MI455X (gfx1250) — matrix operand order repaired and checked by reading, not compiled, not run on hardware
//
#include <hip/hip_runtime.h>
#include <hip/hip_bf16.h>

// Problem constants (fixed by the reference).
#define B_     8
#define S_     4096
#define T_     4096
#define DIM_   512
#define H_     8
#define DH_    64
#define F_     6
#define INNER_ 512
#define SCALE_ 0.125f   // DH^-0.5

typedef __attribute__((ext_vector_type(16))) __bf16 v16bf;
typedef __attribute__((ext_vector_type(8)))  __bf16 v8bf;
typedef __attribute__((ext_vector_type(8)))  float  v8f;

static __device__ __forceinline__ v16bf cmb16(v8bf lo, v8bf hi) {
  v16bf r;
#pragma unroll
  for (int i = 0; i < 8; ++i) { r[i] = lo[i]; r[i + 8] = hi[i]; }
  return r;
}

// ---------------------------------------------------------------------------
// bf16 WMMA GEMM: C[M,N] = A[M,K](fp32) @ W[K,N](fp32) (+bias).
// N, K compile-time; output flavor compile-time (branch-free epilogue).
// Block tile 128x128, BK=32, 8 waves; each wave 32x64 = 2x4 grid of
// v_wmma_f32_16x16x32_bf16. Software-pipelined: global loads for tile k+1
// are in flight (registers) while tile k's WMMAs execute from LDS.
// Requires M%128==0, N%128==0, K%32==0 (true for all uses here).
// ---------------------------------------------------------------------------
template <int NN, int KK, bool OUT_BF16, bool HAS_BIAS>
__global__ __launch_bounds__(256) void gemm_bf16_wmma(
    const float* __restrict__ A, const float* __restrict__ W,
    const float* __restrict__ bias, float* __restrict__ Cf,
    __bf16* __restrict__ Cb)
{
  __shared__ __align__(16) __bf16 As[128 * 40];   // 128 rows x 32 k (pad->40)
  __shared__ __align__(16) __bf16 Bs[128 * 40];   // 128 cols x 32 k (transposed)

  constexpr int NK = KK / 32;

  const int tid   = threadIdx.x;
  const int lane  = tid & 31;
  const int wave  = tid >> 5;
  const int wm    = wave >> 1;     // 0..3 (M direction)
  const int wn    = wave & 1;      // 0..1 (N direction)
  const int mrow  = lane & 15;
  const int khalf = lane >> 4;
  const int m0    = blockIdx.y * 128;
  const int n0    = blockIdx.x * 128;

  // Per-thread staging coordinates.
  const int ar = tid >> 3;             // A row base (0..31), rows ar+i*32
  const int ac = (tid & 7) << 2;       // A col (float4)
  const int wr = tid >> 5;             // W k-row base (0..7), rows wr+i*8
  const int wc = (tid & 31) << 2;      // W n-col (float4)

  const float* Abase = A + (size_t)(m0 + ar) * KK + ac;
  const float* Wbase = W + (size_t)wr * NN + (n0 + wc);

  v8f acc[2][4];
#pragma unroll
  for (int i = 0; i < 2; ++i)
#pragma unroll
    for (int j = 0; j < 4; ++j)
#pragma unroll
      for (int e = 0; e < 8; ++e) acc[i][j][e] = 0.f;

  // Prologue: issue loads for k-tile 0 (8 independent b128 loads).
  float4 areg[4], wreg[4];
#pragma unroll
  for (int i = 0; i < 4; ++i)
    areg[i] = *(const float4*)(Abase + (size_t)i * 32 * KK);
#pragma unroll
  for (int i = 0; i < 4; ++i)
    wreg[i] = *(const float4*)(Wbase + (size_t)i * 8 * NN);

  for (int kt = 0; kt < NK; ++kt) {
    // Convert + store current k-tile registers to LDS.
#pragma unroll
    for (int i = 0; i < 4; ++i) {
      __bf16* dst = &As[(ar + i * 32) * 40 + ac];
      dst[0] = (__bf16)areg[i].x; dst[1] = (__bf16)areg[i].y;
      dst[2] = (__bf16)areg[i].z; dst[3] = (__bf16)areg[i].w;
    }
#pragma unroll
    for (int i = 0; i < 4; ++i) {
      const int r = wr + i * 8;
      Bs[(wc + 0) * 40 + r] = (__bf16)wreg[i].x;
      Bs[(wc + 1) * 40 + r] = (__bf16)wreg[i].y;
      Bs[(wc + 2) * 40 + r] = (__bf16)wreg[i].z;
      Bs[(wc + 3) * 40 + r] = (__bf16)wreg[i].w;
    }
    __syncthreads();

    // Issue next k-tile's global loads (latency hidden under the WMMAs).
    if (kt + 1 < NK) {
      const int k1 = (kt + 1) * 32;
#pragma unroll
      for (int i = 0; i < 4; ++i)
        areg[i] = *(const float4*)(Abase + (size_t)i * 32 * KK + k1);
#pragma unroll
      for (int i = 0; i < 4; ++i)
        wreg[i] = *(const float4*)(Wbase + (size_t)(k1 + i * 8) * NN);
      if (kt + 2 < NK)   // warm L2/WGP$ two tiles ahead (global_prefetch_b8)
        __builtin_prefetch(Abase + (size_t)(kt + 2) * 32, 0, 1);
    }

    // A fragments: lanes 0-15 hold K 0-7 & 16-23, lanes 16-31 K 8-15 & 24-31.
    v16bf afrag[2];
#pragma unroll
    for (int tm = 0; tm < 2; ++tm) {
      int row = wm * 32 + tm * 16 + mrow;
      v8bf lo = *(const v8bf*)&As[row * 40 + khalf * 8];
      v8bf hi = *(const v8bf*)&As[row * 40 + 16 + khalf * 8];
      afrag[tm] = cmb16(lo, hi);
    }
    // B fragments: lanes 0-15 N=lane K 0-7 & 16-23, lanes 16-31 N=lane-16 K 8-15 & 24-31.
    v16bf bfrag[4];
#pragma unroll
    for (int tn = 0; tn < 4; ++tn) {
      int n = wn * 64 + tn * 16 + mrow;
      v8bf lo = *(const v8bf*)&Bs[n * 40 + khalf * 8];
      v8bf hi = *(const v8bf*)&Bs[n * 40 + 16 + khalf * 8];
      bfrag[tn] = cmb16(lo, hi);
    }
#pragma unroll
    for (int tm = 0; tm < 2; ++tm)
#pragma unroll
      for (int tn = 0; tn < 4; ++tn)
        acc[tm][tn] = __builtin_amdgcn_wmma_f32_16x16x32_bf16(
            false, afrag[tm], false, bfrag[tn], (short)0, acc[tm][tn],
            false, false);
    __syncthreads();
  }

  // Epilogue: C/D layout — VGPR r: M = r + 8*(lane>>4), N = lane&15.
#pragma unroll
  for (int tm = 0; tm < 2; ++tm)
#pragma unroll
    for (int tn = 0; tn < 4; ++tn) {
      const int col   = n0 + wn * 64 + tn * 16 + mrow;
      const int rbase = m0 + wm * 32 + tm * 16 + khalf * 8;
      float bv = 0.f;
      if constexpr (HAS_BIAS) bv = bias[col];
#pragma unroll
      for (int r = 0; r < 8; ++r) {
        const float v = acc[tm][tn][r] + bv;
        const size_t off = (size_t)(rbase + r) * NN + col;
        if constexpr (OUT_BF16) Cb[off] = (__bf16)v;
        else                    Cf[off] = v;
      }
    }
}

// ---------------------------------------------------------------------------
// q_filter = filt @ Wqf : [F, INNER] fp32 (tiny: 1.6M MACs)
// ---------------------------------------------------------------------------
__global__ __launch_bounds__(256) void qf_kernel(
    const float* __restrict__ filt, const float* __restrict__ Wqf,
    float* __restrict__ qf)
{
  int idx = blockIdx.x * 256 + threadIdx.x;     // F_*INNER_ = 3072
  if (idx >= F_ * INNER_) return;
  int i = idx % INNER_;
  int f = idx / INNER_;
  float acc = 0.f;
  for (int k = 0; k < DIM_; ++k)
    acc += filt[f * DIM_ + k] * Wqf[(size_t)k * INNER_ + i];
  qf[idx] = acc;
}

// ---------------------------------------------------------------------------
// dots1[b,h,f,s] = SCALE * qf[f, h*64+:64] . k1[b,s, h*64+:64]   (k1 = kv1 lo)
// ---------------------------------------------------------------------------
__global__ __launch_bounds__(256) void dots1_kernel(
    const float* __restrict__ qf, const __bf16* __restrict__ kv1,
    float* __restrict__ dots)
{
  int idx = blockIdx.x * 256 + threadIdx.x;     // (((b*H)+h)*F+f)*S+s
  int s = idx & (S_ - 1);
  int r = idx >> 12;                            // S_ = 4096
  int f = r % F_; r /= F_;
  int h = r % H_;
  int b = r / H_;
  const float*  q = qf + f * INNER_ + h * DH_;
  const __bf16* k = kv1 + ((size_t)(b * S_ + s)) * (2 * INNER_) + h * DH_;
  float acc = 0.f;
#pragma unroll
  for (int d = 0; d < DH_; ++d) acc += q[d] * (float)k[d];
  dots[idx] = acc * SCALE_;
}

// ---------------------------------------------------------------------------
// Per (b,h,f): softmax over S, then out1[b,h,f,d] = sum_s attn[s]*v1[b,s,h,d]
// ---------------------------------------------------------------------------
__global__ __launch_bounds__(256) void soft1_kernel(
    const float* __restrict__ dots, const __bf16* __restrict__ kv1,
    float* __restrict__ out1)
{
  __shared__ float wsm[S_];      // 16 KB of attention weights
  __shared__ float red[256];
  int bhf = blockIdx.x;          // (b*H + h)*F + f
  int bh  = bhf / F_;
  int h   = bh % H_;
  int b   = bh / H_;
  const float*  drow = dots + (size_t)bhf * S_;
  const __bf16* v1   = kv1 + (size_t)b * S_ * (2 * INNER_) + INNER_ + h * DH_;
  int t = threadIdx.x;

  float mx = -INFINITY;
  for (int s = t; s < S_; s += 256) mx = fmaxf(mx, drow[s]);
  red[t] = mx; __syncthreads();
  for (int o = 128; o > 0; o >>= 1) {
    if (t < o) red[t] = fmaxf(red[t], red[t + o]);
    __syncthreads();
  }
  mx = red[0]; __syncthreads();

  float sm = 0.f;
  for (int s = t; s < S_; s += 256) {
    float e = __expf(drow[s] - mx);
    wsm[s] = e;
    sm += e;
  }
  red[t] = sm; __syncthreads();
  for (int o = 128; o > 0; o >>= 1) {
    if (t < o) red[t] += red[t + o];
    __syncthreads();
  }
  float inv = 1.f / red[0];
  __syncthreads();

  // threads = 64 (d) x 4 (s-group) partial accumulation over v1
  int d  = t & 63;
  int sg = t >> 6;
  float acc = 0.f;
  for (int s = sg; s < S_; s += 4)
    acc += wsm[s] * (float)v1[(size_t)s * (2 * INNER_) + d];
  red[t] = acc; __syncthreads();
  if (t < 64) {
    float r = red[t] + red[t + 64] + red[t + 128] + red[t + 192];
    out1[(size_t)bhf * DH_ + t] = r * inv;
  }
}

// ---------------------------------------------------------------------------
// out1m[b,f,j] = bo1[j] + sum_i merge(out1)[b,f,i] * Wo1[i,j]
// ---------------------------------------------------------------------------
__global__ __launch_bounds__(256) void out1proj_kernel(
    const float* __restrict__ out1, const float* __restrict__ Wo1,
    const float* __restrict__ bo1, float* __restrict__ out1m)
{
  int idx = blockIdx.x * 256 + threadIdx.x;     // B_*F_*DIM_ = 24576
  int j = idx % DIM_;
  int f = (idx / DIM_) % F_;
  int b = idx / (DIM_ * F_);
  float acc = bo1[j];
  for (int i = 0; i < INNER_; ++i) {
    int h = i >> 6, d = i & 63;
    float m = out1[(((size_t)b * H_ + h) * F_ + f) * DH_ + d];
    acc += m * Wo1[(size_t)i * DIM_ + j];
  }
  out1m[idx] = acc;
}

// ---------------------------------------------------------------------------
// kv2[b,f,j] = sum_i out1m[b,f,i] * Wkv2[i,j]   (j < 2*INNER)
// ---------------------------------------------------------------------------
__global__ __launch_bounds__(256) void kv2_kernel(
    const float* __restrict__ out1m, const float* __restrict__ Wkv2,
    float* __restrict__ kv2)
{
  int idx = blockIdx.x * 256 + threadIdx.x;     // B_*F_*2*INNER_ = 49152
  int j = idx % (2 * INNER_);
  int f = (idx / (2 * INNER_)) % F_;
  int b = idx / (2 * INNER_ * F_);
  float acc = 0.f;
  for (int i = 0; i < DIM_; ++i)
    acc += out1m[((size_t)b * F_ + f) * DIM_ + i] * Wkv2[(size_t)i * (2 * INNER_) + j];
  kv2[idx] = acc;
}

// ---------------------------------------------------------------------------
// Stage 2: per (b,h,t): 6 dots vs k2, softmax over F=6, out = attn @ v2.
// ---------------------------------------------------------------------------
__global__ __launch_bounds__(256) void attn2_kernel(
    const __bf16* __restrict__ qt, const float* __restrict__ kv2,
    float* __restrict__ out2)
{
  int idx = blockIdx.x * 256 + threadIdx.x;     // (b*H + h)*T + t
  int t  = idx & (T_ - 1);
  int bh = idx >> 12;
  int h  = bh % H_;
  int b  = bh / H_;
  const __bf16* q  = qt + (size_t)(b * T_ + t) * INNER_ + h * DH_;
  const float*  k2 = kv2 + (size_t)b * F_ * (2 * INNER_) + h * DH_;
  const float*  v2 = k2 + INNER_;

  float qr[DH_];
#pragma unroll
  for (int d = 0; d < DH_; ++d) qr[d] = (float)q[d];

  float dots[F_];
  float mx = -INFINITY;
#pragma unroll
  for (int f = 0; f < F_; ++f) {
    const float* kf = k2 + (size_t)f * (2 * INNER_);
    float a = 0.f;
#pragma unroll
    for (int d = 0; d < DH_; ++d) a += qr[d] * kf[d];
    a *= SCALE_;
    dots[f] = a;
    mx = fmaxf(mx, a);
  }
  float se = 0.f;
#pragma unroll
  for (int f = 0; f < F_; ++f) { dots[f] = __expf(dots[f] - mx); se += dots[f]; }
  float inv = 1.f / se;

  float* o = out2 + (size_t)(b * T_ + t) * INNER_ + h * DH_;
#pragma unroll
  for (int d = 0; d < DH_; ++d) {
    float a = 0.f;
#pragma unroll
    for (int f = 0; f < F_; ++f) a += dots[f] * v2[(size_t)f * (2 * INNER_) + d];
    o[d] = a * inv;
  }
}

// ---------------------------------------------------------------------------
// Host-side orchestration. Inputs (setup_inputs order):
// 0 src, 1 src_pos_emb (unused), 2 tgt, 3 tgt_pos_emb (unused), 4 filt,
// 5 Wqf, 6 Wqt, 7 Wkv1, 8 Wkv2, 9 Wo1, 10 bo1, 11 Wo2, 12 bo2.
// Workspace layout (~174.5 MB total, all 256B-aligned):
//   kv1 bf16 64MB | qt bf16 32MB | out2f fp32 64MB | dots1 fp32 6MB |
//   qf 12KB | out1 96KB | out1m 96KB | kv2 192KB
// ---------------------------------------------------------------------------
extern "C" void kernel_launch(void* const* d_in, const int* in_sizes, int n_in,
                              void* d_out, int out_size, void* d_ws, size_t ws_size,
                              hipStream_t stream)
{
  (void)in_sizes; (void)n_in; (void)out_size; (void)ws_size;

  const float* src  = (const float*)d_in[0];
  const float* tgt  = (const float*)d_in[2];
  const float* filt = (const float*)d_in[4];
  const float* Wqf  = (const float*)d_in[5];
  const float* Wqt  = (const float*)d_in[6];
  const float* Wkv1 = (const float*)d_in[7];
  const float* Wkv2 = (const float*)d_in[8];
  const float* Wo1  = (const float*)d_in[9];
  const float* bo1  = (const float*)d_in[10];
  const float* Wo2  = (const float*)d_in[11];
  const float* bo2  = (const float*)d_in[12];
  float* out = (float*)d_out;

  char* ws = (char*)d_ws;
  __bf16* kv1   = (__bf16*)(ws);                  //  67,108,864 B
  __bf16* qt    = (__bf16*)(ws +  67108864ull);   //  33,554,432 B
  float*  out2f = (float*) (ws + 100663296ull);   //  67,108,864 B
  float*  dots1 = (float*) (ws + 167772160ull);   //   6,291,456 B
  float*  qf    = (float*) (ws + 174063616ull);   //      12,288 B
  float*  out1  = (float*) (ws + 174075904ull);   //      98,304 B
  float*  out1m = (float*) (ws + 174174208ull);   //      98,304 B
  float*  kv2   = (float*) (ws + 174272512ull);   //     196,608 B

  dim3 blk(256);

  // kv1 = src @ Wkv1 -> bf16    (M=32768, N=1024, K=512)
  gemm_bf16_wmma<1024, 512, true, false><<<dim3(8, 256), blk, 0, stream>>>(
      src, Wkv1, nullptr, nullptr, kv1);
  // qt = tgt @ Wqt -> bf16      (M=32768, N=512, K=512)
  gemm_bf16_wmma<512, 512, true, false><<<dim3(4, 256), blk, 0, stream>>>(
      tgt, Wqt, nullptr, nullptr, qt);

  qf_kernel<<<(F_ * INNER_ + 255) / 256, blk, 0, stream>>>(filt, Wqf, qf);
  dots1_kernel<<<(B_ * H_ * F_ * S_) / 256, blk, 0, stream>>>(qf, kv1, dots1);
  soft1_kernel<<<B_ * H_ * F_, blk, 0, stream>>>(dots1, kv1, out1);
  out1proj_kernel<<<(B_ * F_ * DIM_) / 256, blk, 0, stream>>>(out1, Wo1, bo1, out1m);
  kv2_kernel<<<(B_ * F_ * 2 * INNER_) / 256, blk, 0, stream>>>(out1m, Wkv2, kv2);
  attn2_kernel<<<(B_ * H_ * T_) / 256, blk, 0, stream>>>(qt, kv2, out2f);

  // out = out2f @ Wo2 + bo2 -> fp32   (M=32768, N=512, K=512)
  gemm_bf16_wmma<512, 512, false, true><<<dim3(4, 256), blk, 0, stream>>>(
      out2f, Wo2, bo2, out, nullptr);
}
